// MultiHeadAttention_16063177687643
// MI455X (gfx1250) — compile-verified, matrix operand order checked by reading, not run on hardware
//
#include <hip/hip_runtime.h>
#include <hip/hip_bf16.h>

typedef _Float16 half_t;
typedef __attribute__((ext_vector_type(16))) _Float16 v16h;
typedef __attribute__((ext_vector_type(8)))  _Float16 v8h;
typedef __attribute__((ext_vector_type(8)))  float    v8f;
typedef __attribute__((ext_vector_type(4)))  float    v4f;

#define NB    2
#define SEQ   2048
#define DM    768
#define NH    12
#define DHEAD 64
#define LN_EPS 1e-5f

// padded LDS row stride for score tile (floats)
#define SROW  2056

// ---- workspace layout (bytes) ----
#define OFF_WQT  ((size_t)0)            // 768*768*2
#define OFF_WKT  ((size_t)1179648)
#define OFF_WVT  ((size_t)2359296)
#define OFF_WOT  ((size_t)3538944)
#define OFF_Q    ((size_t)4718592)      // B*H*S*64 f16 = 6291456 B
#define OFF_K    ((size_t)11010048)
#define OFF_V    ((size_t)17301504)
#define OFF_VT   ((size_t)23592960)
#define OFF_CTX  ((size_t)29884416)     // B*S*768 f16
#define OFF_OUTP ((size_t)36175872)     // B*S*768 f32

#define WMMA_F16(a,b,c) \
  __builtin_amdgcn_wmma_f32_16x16x32_f16(false,(a),false,(b),(short)0,(c),false,false)

static __device__ __forceinline__ v16h combine8(v8h lo, v8h hi) {
  v16h r;
#pragma unroll
  for (int j = 0; j < 8; ++j) { r[j] = lo[j]; r[j + 8] = hi[j]; }
  return r;
}

// contiguous f16 fragment load: elements {khalf..khalf+7, 16+khalf..16+khalf+7}
static __device__ __forceinline__ v16h load_frag_f16(const half_t* p, int khalf) {
  v8h lo = *(const v8h*)(p + khalf);
  v8h hi = *(const v8h*)(p + 16 + khalf);
  return combine8(lo, hi);
}

// fp32 -> f16 fragment (p already offset by khalf)
static __device__ __forceinline__ v16h cvt_frag_f32(const float* p) {
  v4f a0 = *(const v4f*)(p);
  v4f a1 = *(const v4f*)(p + 4);
  v4f a2 = *(const v4f*)(p + 16);
  v4f a3 = *(const v4f*)(p + 20);
  v16h r;
#pragma unroll
  for (int j = 0; j < 4; ++j) {
    r[j]      = (half_t)a0[j];
    r[4 + j]  = (half_t)a1[j];
    r[8 + j]  = (half_t)a2[j];
    r[12 + j] = (half_t)a3[j];
  }
  return r;
}

// ---------------- weight transpose + fp32->f16  (W[k][n] -> WT[n][k]) --------
__global__ void k_wtrans(const float* __restrict__ W, half_t* __restrict__ WT) {
  __shared__ float tile[32][33];
  const int bx = blockIdx.x % (DM / 32);      // n tile
  const int by = blockIdx.x / (DM / 32);      // k tile
  const int n0 = bx * 32, k0 = by * 32;
  const int tx = threadIdx.x & 31, ty = threadIdx.x >> 5;   // 32 x 8
#pragma unroll
  for (int i = 0; i < 4; ++i) {
    int k = ty + i * 8;
    tile[k][tx] = W[(k0 + k) * DM + n0 + tx];
  }
  __syncthreads();
#pragma unroll
  for (int i = 0; i < 4; ++i) {
    int n = ty + i * 8;
    WT[(size_t)(n0 + n) * DM + k0 + tx] = (half_t)tile[tx][n];
  }
}

// ---------------- QKV projection GEMM:  out[b][h][s][d] = (X @ W + bias) -----
// one wave -> 16(M) x 64(N) tile, K=768 in steps of 32
__global__ void __launch_bounds__(256) k_proj(const float* __restrict__ X,
                                              const half_t* __restrict__ WT,
                                              const float* __restrict__ bias,
                                              half_t* __restrict__ out) {
  const int wave  = (blockIdx.x * blockDim.x + threadIdx.x) >> 5;
  const int lane  = threadIdx.x & 31;
  const int mt    = wave / (DM / 64);         // 0..255
  const int h     = wave % (DM / 64);         // 0..11
  const int lm    = lane & 15;
  const int khalf = (lane >> 4) << 3;
  const int arow  = mt * 16 + lm;

  v8f acc[4] = {v8f{}, v8f{}, v8f{}, v8f{}};
  for (int k0 = 0; k0 < DM; k0 += 32) {
    v16h a = cvt_frag_f32(X + (size_t)arow * DM + k0 + khalf);
#pragma unroll
    for (int sub = 0; sub < 4; ++sub) {
      const int n = h * 64 + sub * 16 + lm;
      v16h bf = load_frag_f16(WT + (size_t)n * DM + k0, khalf);
      acc[sub] = WMMA_F16(a, bf, acc[sub]);
    }
  }
  const int mrow0 = (lane >> 4) << 3;
#pragma unroll
  for (int sub = 0; sub < 4; ++sub) {
    const int d  = sub * 16 + lm;
    const float bv = bias[h * 64 + d];
#pragma unroll
    for (int r = 0; r < 8; ++r) {
      const int srow = mt * 16 + r + mrow0;     // tile never crosses batch
      const int bb = srow >> 11, ss = srow & (SEQ - 1);
      out[(((size_t)(bb * NH + h) * SEQ) + ss) * DHEAD + d] = (half_t)(acc[sub][r] + bv);
    }
  }
}

// ---------------- v[bh][s][64] -> vT[bh][64][s] ------------------------------
__global__ void k_vtrans(const half_t* __restrict__ v, half_t* __restrict__ vT) {
  __shared__ half_t tile[64][72];
  const int bh = blockIdx.x / (SEQ / 64);
  const int s0 = (blockIdx.x % (SEQ / 64)) * 64;
  const int c  = threadIdx.x & 63, r0 = threadIdx.x >> 6;
#pragma unroll
  for (int i = 0; i < 16; ++i) {
    int s = r0 + i * 4;
    tile[s][c] = v[((size_t)bh * SEQ + s0 + s) * DHEAD + c];
  }
  __syncthreads();
#pragma unroll
  for (int i = 0; i < 16; ++i) {
    int d = r0 + i * 4;
    vT[((size_t)bh * DHEAD + d) * SEQ + s0 + c] = tile[c][d];
  }
}

// ---------------- fused attention: scores -> mask -> softmax -> attn, ctx ----
// one block per (b, h, 16 q-rows). dynamic LDS:
//   scb[16][SROW] f32 | red[256] f32 | part[8][256] f32
__global__ void __launch_bounds__(256) k_attn(const half_t* __restrict__ qf,
                                              const half_t* __restrict__ kf,
                                              const half_t* __restrict__ vT,
                                              const unsigned char* __restrict__ mask,
                                              float* __restrict__ attn_out,
                                              half_t* __restrict__ ctx) {
  extern __shared__ float smem[];
  float* scb  = smem;
  float* red  = smem + 16 * SROW;
  float* part = red + 256;

  const int bh = blockIdx.x / (SEQ / 16);
  const int qt = blockIdx.x % (SEQ / 16);
  const int b = bh / NH, h = bh % NH;
  const int tid = threadIdx.x;
  const int w = tid >> 5, lane = tid & 31;
  const int lm = lane & 15;
  const int khalf = (lane >> 4) << 3;
  const int mrow0 = (lane >> 4) << 3;

  // Q fragments for this wave's 16 rows (K = 64 -> two K=32 fragments)
  const half_t* qp = qf + ((size_t)bh * SEQ + qt * 16 + lm) * DHEAD;
  const v16h qa0 = load_frag_f16(qp, khalf);
  const v16h qa1 = load_frag_f16(qp + 32, khalf);

  // ---- phase 1: raw scaled scores for this wave's 256-key stripe ----
  // (mask is applied in phase 2 where its reads are fully coalesced)
  const int kbase = w * 256;
  for (int kt = 0; kt < 16; ++kt) {
    const int key = kbase + kt * 16 + lm;
    const half_t* kp = kf + ((size_t)bh * SEQ + key) * DHEAD;
    v16h b0 = load_frag_f16(kp, khalf);
    v16h b1 = load_frag_f16(kp + 32, khalf);
    v8f acc = {};
    acc = WMMA_F16(qa0, b0, acc);
    acc = WMMA_F16(qa1, b1, acc);
    const int col = kbase + kt * 16 + lm;
#pragma unroll
    for (int r = 0; r < 8; ++r) {
      scb[(r + mrow0) * SROW + col] = acc[r] * 0.125f;   // 1/sqrt(64)
    }
  }
  __syncthreads();

  // ---- phase 2: mask + softmax over each of the 16 rows ----
  {
    const int rrow = tid >> 4, seg = tid & 15;          // 16 threads / row
    float* rp = scb + rrow * SROW + seg * 128;
    // coalesced mask bytes for this thread's 128 contiguous columns
    const uint4* mp4 = (const uint4*)(mask + ((size_t)b * SEQ + qt * 16 + rrow) * SEQ
                                           + seg * 128);
    float mx = -3.0e38f;
#pragma unroll
    for (int blk = 0; blk < 8; ++blk) {
      const uint4 mw = mp4[blk];
      const unsigned int wd[4] = {mw.x, mw.y, mw.z, mw.w};
#pragma unroll
      for (int q = 0; q < 4; ++q) {
#pragma unroll
        for (int j = 0; j < 4; ++j) {
          const int c = blk * 16 + q * 4 + j;
          float sv = rp[c];
          if ((wd[q] >> (8 * j)) & 0xFF) sv = -1e9f;    // True => masked
          rp[c] = sv;
          mx = fmaxf(mx, sv);
        }
      }
    }
    red[rrow * 16 + seg] = mx;
    __syncthreads();
    float rowmax = red[rrow * 16];
#pragma unroll
    for (int j = 1; j < 16; ++j) rowmax = fmaxf(rowmax, red[rrow * 16 + j]);
    __syncthreads();
    float ls = 0.f;
#pragma unroll 8
    for (int c = 0; c < 128; ++c) {
      float e = __expf(rp[c] - rowmax);
      rp[c] = e;
      ls += e;
    }
    red[rrow * 16 + seg] = ls;
    __syncthreads();
    float rsum = 0.f;
#pragma unroll
    for (int j = 0; j < 16; ++j) rsum += red[rrow * 16 + j];
    const float rinv = 1.0f / rsum;
#pragma unroll 8
    for (int c = 0; c < 128; ++c) rp[c] *= rinv;
  }
  __syncthreads();

  // ---- phase 3: stream attn tile to global (coalesced, non-temporal:
  //      written once, never re-read -> keep 402 MB out of L2) ----
  float* aout = attn_out + ((size_t)bh * SEQ + qt * 16) * SEQ;
  for (int i = tid; i < 16 * SEQ; i += 256) {
    const int m = i >> 11, c = i & (SEQ - 1);
    __builtin_nontemporal_store(scb[m * SROW + c], &aout[(size_t)m * SEQ + c]);
  }

  // ---- phase 4: ctx = attn @ V   (8 waves = 4 d-tiles x 2 K-halves) ----
  const int nt2 = w & 3, kpart = w >> 2;
  const int dcol = nt2 * 16 + lm;
  const half_t* vp = vT + ((size_t)bh * DHEAD + dcol) * SEQ + kpart * 1024;
  v8f cacc = {};
  for (int k0 = 0; k0 < 1024; k0 += 32) {
    const float* ap = scb + lm * SROW + kpart * 1024 + k0 + khalf;
    v16h af;
#pragma unroll
    for (int j = 0; j < 8; ++j) {
      af[j]     = (half_t)ap[j];
      af[j + 8] = (half_t)ap[16 + j];
    }
    v16h bf = load_frag_f16(vp + k0, khalf);
    cacc = WMMA_F16(af, bf, cacc);
  }
  float* pw = part + w * 256;
#pragma unroll
  for (int r = 0; r < 8; ++r) pw[(r + mrow0) * 16 + lm] = cacc[r];
  __syncthreads();

  half_t* cp = ctx + ((size_t)(b * SEQ + qt * 16)) * DM + h * 64;
  for (int i = tid; i < 1024; i += 256) {
    const int m = i >> 6, d = i & 63;
    const int n2 = d >> 4, dl = d & 15;
    const float vsum = part[n2 * 256 + m * 16 + dl] + part[(4 + n2) * 256 + m * 16 + dl];
    cp[(size_t)m * DM + d] = (half_t)vsum;
  }
}

// ---------------- output projection + residual -------------------------------
__global__ void __launch_bounds__(256) k_outproj(const half_t* __restrict__ ctx,
                                                 const half_t* __restrict__ WoT,
                                                 const float* __restrict__ bo,
                                                 const float* __restrict__ Qres,
                                                 float* __restrict__ outp) {
  const int wave  = (blockIdx.x * blockDim.x + threadIdx.x) >> 5;
  const int lane  = threadIdx.x & 31;
  const int mt    = wave / (DM / 64);
  const int nt    = wave % (DM / 64);
  const int lm    = lane & 15;
  const int khalf = (lane >> 4) << 3;
  const int arow  = mt * 16 + lm;

  v8f acc[4] = {v8f{}, v8f{}, v8f{}, v8f{}};
  for (int k0 = 0; k0 < DM; k0 += 32) {
    v16h a = load_frag_f16(ctx + (size_t)arow * DM + k0, khalf);
#pragma unroll
    for (int sub = 0; sub < 4; ++sub) {
      const int n = nt * 64 + sub * 16 + lm;
      v16h bf = load_frag_f16(WoT + (size_t)n * DM + k0, khalf);
      acc[sub] = WMMA_F16(a, bf, acc[sub]);
    }
  }
  const int mrow0 = (lane >> 4) << 3;
#pragma unroll
  for (int sub = 0; sub < 4; ++sub) {
    const int n = nt * 64 + sub * 16 + lm;
    const float bv = bo[n];
#pragma unroll
    for (int r = 0; r < 8; ++r) {
      const int srow = mt * 16 + r + mrow0;
      outp[(size_t)srow * DM + n] = acc[sub][r] + bv + Qres[(size_t)srow * DM + n];
    }
  }
}

// ---------------- layernorm ---------------------------------------------------
__global__ void k_ln(const float* __restrict__ x, const float* __restrict__ gamma,
                     const float* __restrict__ beta, float* __restrict__ y) {
  __shared__ float rbuf[256];
  const int row = blockIdx.x;
  const int tid = threadIdx.x;
  const float* xp = x + (size_t)row * DM;
  float lv[3], s = 0.f;
#pragma unroll
  for (int i = 0; i < 3; ++i) { lv[i] = xp[tid + i * 256]; s += lv[i]; }
  rbuf[tid] = s; __syncthreads();
  for (int off = 128; off > 0; off >>= 1) {
    if (tid < off) rbuf[tid] += rbuf[tid + off];
    __syncthreads();
  }
  const float mean = rbuf[0] * (1.0f / DM);
  __syncthreads();
  float sq = 0.f;
#pragma unroll
  for (int i = 0; i < 3; ++i) { float d = lv[i] - mean; sq += d * d; }
  rbuf[tid] = sq; __syncthreads();
  for (int off = 128; off > 0; off >>= 1) {
    if (tid < off) rbuf[tid] += rbuf[tid + off];
    __syncthreads();
  }
  const float rstd = rsqrtf(rbuf[0] * (1.0f / DM) + LN_EPS);
#pragma unroll
  for (int i = 0; i < 3; ++i) {
    const int c = tid + i * 256;
    const float val = gamma[c] * (lv[i] - mean) * rstd + beta[c];
    __builtin_nontemporal_store(val, &y[(size_t)row * DM + c]);
  }
}

extern "C" void kernel_launch(void* const* d_in, const int* in_sizes, int n_in,
                              void* d_out, int out_size, void* d_ws, size_t ws_size,
                              hipStream_t stream) {
  (void)in_sizes; (void)n_in; (void)out_size; (void)ws_size;
  const float* Q    = (const float*)d_in[0];
  const float* K    = (const float*)d_in[1];
  const float* V    = (const float*)d_in[2];
  const unsigned char* mask = (const unsigned char*)d_in[3];
  const float* Wq   = (const float*)d_in[4];
  const float* bq   = (const float*)d_in[5];
  const float* Wk   = (const float*)d_in[6];
  const float* bk   = (const float*)d_in[7];
  const float* Wv   = (const float*)d_in[8];
  const float* bv   = (const float*)d_in[9];
  const float* Wo   = (const float*)d_in[10];
  const float* bo   = (const float*)d_in[11];
  const float* gamma= (const float*)d_in[12];
  const float* beta = (const float*)d_in[13];

  char* ws = (char*)d_ws;
  half_t* WqT = (half_t*)(ws + OFF_WQT);
  half_t* WkT = (half_t*)(ws + OFF_WKT);
  half_t* WvT = (half_t*)(ws + OFF_WVT);
  half_t* WoT = (half_t*)(ws + OFF_WOT);
  half_t* qh  = (half_t*)(ws + OFF_Q);
  half_t* kh  = (half_t*)(ws + OFF_K);
  half_t* vh  = (half_t*)(ws + OFF_V);
  half_t* vT  = (half_t*)(ws + OFF_VT);
  half_t* ctx = (half_t*)(ws + OFF_CTX);
  float*  outp= (float*)(ws + OFF_OUTP);

  float* y    = (float*)d_out;
  float* attn = (float*)d_out + (size_t)NB * SEQ * DM;

  const int wt_blocks  = (DM / 32) * (DM / 32);             // 576
  k_wtrans<<<wt_blocks, 256, 0, stream>>>(Wq, WqT);
  k_wtrans<<<wt_blocks, 256, 0, stream>>>(Wk, WkT);
  k_wtrans<<<wt_blocks, 256, 0, stream>>>(Wv, WvT);
  k_wtrans<<<wt_blocks, 256, 0, stream>>>(Wo, WoT);

  const int proj_blocks = (NB * SEQ / 16) * (DM / 64) / 8;  // 384
  k_proj<<<proj_blocks, 256, 0, stream>>>(Q, WqT, bq, qh);
  k_proj<<<proj_blocks, 256, 0, stream>>>(K, WkT, bk, kh);
  k_proj<<<proj_blocks, 256, 0, stream>>>(V, WvT, bv, vh);

  k_vtrans<<<NB * NH * (SEQ / 64), 256, 0, stream>>>(vh, vT);

  const size_t attn_lds = (size_t)(16 * SROW + 256 + 8 * 256) * sizeof(float); // ~140.8 KB
  k_attn<<<NB * NH * (SEQ / 16), 256, attn_lds, stream>>>(qh, kh, vT, mask, attn, ctx);

  k_outproj<<<proj_blocks, 256, 0, stream>>>(ctx, WoT, bo, Q, outp);

  k_ln<<<NB * SEQ, 256, 0, stream>>>(outp, gamma, beta, y);
}
